// MultiScaleRetention_78065325572194
// MI455X (gfx1250) — matrix operand order repaired and checked by reading, not compiled, not run on hardware
//
#include <hip/hip_runtime.h>
#include <hip/hip_bf16.h>

// ---- MI455X / gfx1250: wave32, WMMA bf16 path ----
typedef __bf16 bf16;
typedef __attribute__((ext_vector_type(16))) __bf16 bf16x16;
typedef __attribute__((ext_vector_type(8)))  __bf16 bf16x8;
typedef __attribute__((ext_vector_type(4)))  __bf16 bf16x4;
typedef __attribute__((ext_vector_type(4)))  float  f32x4;
typedef __attribute__((ext_vector_type(8)))  float  f32x8;

#define WMMA_BF16(A, B, C) \
  __builtin_amdgcn_wmma_f32_16x16x32_bf16(false, (A), false, (B), (short)0, (C), false, false)

// Problem constants (match reference)
#define BB 2
#define TT 2048
#define DD 1024
#define HH 16
#define HD 64
#define NTOK (BB * TT)          // 4096
#define EPS_GN 1e-5f

// ---------------------------------------------------------------------------
// Kernel 1: fp32 -> bf16 downcast of x (4M elems) and the 4 weight matrices
// (4 x 1M elems), 4 elements per thread (vectorized b128 load / b64 store).
// ---------------------------------------------------------------------------
__global__ __launch_bounds__(256) void msr_convert_kernel(
    const float* __restrict__ x,
    const float* __restrict__ wq, const float* __restrict__ wk,
    const float* __restrict__ wv, const float* __restrict__ wg,
    bf16* __restrict__ xb, bf16* __restrict__ wb)
{
  const size_t i4 = ((size_t)blockIdx.x * 256 + threadIdx.x) * 4;
  const size_t NX = (size_t)NTOK * DD;              // 4M
  const float* src;
  bf16* dst;
  if (i4 < NX) {
    src = x + i4;
    dst = xb + i4;
  } else {
    const size_t j = i4 - NX;                       // 0..4M-1
    const int    m = (int)(j >> 20);                // which weight matrix
    const size_t o = j & ((1u << 20) - 1);          // offset within (out,in)
    src = ((m == 0) ? wq : (m == 1) ? wk : (m == 2) ? wv : wg) + o;
    dst = wb + j;
  }
  const f32x4 v = *(const f32x4*)src;
  bf16x4 r;
#pragma unroll
  for (int q = 0; q < 4; ++q) r[q] = (bf16)v[q];
  *(bf16x4*)dst = r;
}

// ---------------------------------------------------------------------------
// Kernel 2: fused QKVG projection GEMM.  Y[n, e] = sum_d x[n,d] * W[e,d] + b.
// NT GEMM (both operands K-contiguous) -> WMMA fragment loads are contiguous.
// Wave computes 32x64 (2x4 WMMA grid).  All 6 fragments of a K-step are
// loaded into distinct registers BEFORE the 8 WMMAs so the scheduler can
// clause the loads and overlap the next iteration's loads with compute.
// ---------------------------------------------------------------------------
__global__ __launch_bounds__(128) void msr_qkvg_gemm_kernel(
    const bf16* __restrict__ xb, const bf16* __restrict__ wb,
    const float* __restrict__ bq, const float* __restrict__ bk,
    const float* __restrict__ bv, const float* __restrict__ bg,
    bf16* __restrict__ Qb, bf16* __restrict__ Kb, bf16* __restrict__ Vt,
    float* __restrict__ G)
{
  const int lane  = threadIdx.x & 31;
  const int wave  = threadIdx.x >> 5;
  const int khalf = lane >> 4;          // 0/1: which K-half this lane holds
  const int l16   = lane & 15;
  const int m0 = blockIdx.x * 128 + wave * 32;  // token rows (2 sub-tiles)
  const int n0 = blockIdx.y * 64;               // output cols (0..4095)

  f32x8 acc[2][4] = {};
  const bf16* arow0 = xb + (size_t)(m0 + l16) * DD + khalf * 16;
  const bf16* arow1 = arow0 + (size_t)16 * DD;
  const bf16* brow  = wb + (size_t)(n0 + l16) * DD + khalf * 16;

  for (int kk = 0; kk < DD; kk += 32) {
    // ---- load ALL fragments for this K-step first (distinct registers) ----
    bf16x16 a[2];
    bf16x16 bfr[4];
    {
      bf16x8 a00 = *(const bf16x8*)(arow0 + kk);
      bf16x8 a01 = *(const bf16x8*)(arow0 + kk + 8);
      bf16x8 a10 = *(const bf16x8*)(arow1 + kk);
      bf16x8 a11 = *(const bf16x8*)(arow1 + kk + 8);
#pragma unroll
      for (int q = 0; q < 8; ++q) {
        a[0][q] = a00[q]; a[0][q + 8] = a01[q];
        a[1][q] = a10[q]; a[1][q + 8] = a11[q];
      }
    }
#pragma unroll
    for (int t = 0; t < 4; ++t)
      bfr[t] = *(const bf16x16*)(brow + (size_t)t * 16 * DD + kk);

    // ---- 8 WMMAs: each B-frag reused 2x, each A-frag 4x ----
#pragma unroll
    for (int t = 0; t < 4; ++t) {
      acc[0][t] = WMMA_BF16(a[0], bfr[t], acc[0][t]);
      acc[1][t] = WMMA_BF16(a[1], bfr[t], acc[1][t]);
    }
  }

  // Epilogue: bias + scatter to Q/K/Vt/G.  mat is uniform per block.
  const int mat = n0 >> 10;
  const float* bias = (mat == 0) ? bq : (mat == 1) ? bk : (mat == 2) ? bv : bg;
#pragma unroll
  for (int mi = 0; mi < 2; ++mi) {
#pragma unroll
    for (int t = 0; t < 4; ++t) {
      const int e  = n0 + t * 16 + l16;
      const int ec = e & (DD - 1);
      const float bia = bias[ec];
      const int h = ec >> 6, hd = ec & 63;
#pragma unroll
      for (int r = 0; r < 8; ++r) {
        const int row = m0 + mi * 16 + r + khalf * 8;   // token index
        const int b = row >> 11, tt_ = row & (TT - 1);
        const int bh = b * HH + h;
        const float v = acc[mi][t][r] + bia;
        if (mat == 0)
          Qb[((size_t)bh * TT + tt_) * HD + hd] = (bf16)(v * 0.125f); // 1/sqrt(64)
        else if (mat == 1)
          Kb[((size_t)bh * TT + tt_) * HD + hd] = (bf16)v;
        else if (mat == 2)
          Vt[((size_t)bh * HD + hd) * TT + tt_]  = (bf16)v;           // transposed
        else
          G[(size_t)row * DD + ec] = v;
      }
    }
  }
}

// ---------------------------------------------------------------------------
// Kernel 3: flash attention + gate + per-(token,head) GroupNorm, fused.
// grid = (T/64, B*H); block = 128 threads = 4 waves; wave owns 16 query rows.
// Inner loop: 32 keys/iter -> 4 WMMA (scores) + LDS C->A relayout + 4 WMMA (PV).
// Fragments loaded before compute in each phase for load/WMMA overlap.
// ---------------------------------------------------------------------------
__global__ __launch_bounds__(128) void msr_attn_gn_kernel(
    const bf16* __restrict__ Qb, const bf16* __restrict__ Kb,
    const bf16* __restrict__ Vt, const float* __restrict__ G,
    const float* __restrict__ gamma, const float* __restrict__ beta,
    float* __restrict__ out)
{
  __shared__ __align__(32) bf16 lds_p[4][16 * 32];  // per-wave P staging

  const int lane  = threadIdx.x & 31;
  const int wave  = threadIdx.x >> 5;
  const int khalf = lane >> 4;
  const int l16   = lane & 15;
  const int bh = blockIdx.y;                 // b*16 + h
  const int b  = bh >> 4, h = bh & 15;
  const int t0 = blockIdx.x * 64 + wave * 16;

  // Q fragments (two 16x32 K-chunks over head dim 64), pre-scaled by 1/8
  bf16x16 aq[2];
#pragma unroll
  for (int c = 0; c < 2; ++c) {
    const bf16* p = Qb + ((size_t)bh * TT + t0 + l16) * HD + c * 32 + khalf * 16;
    bf16x8 q0 = *(const bf16x8*)p;
    bf16x8 q1 = *(const bf16x8*)(p + 8);
#pragma unroll
    for (int q = 0; q < 8; ++q) { aq[c][q] = q0[q]; aq[c][q + 8] = q1[q]; }
  }

  f32x8 o[4] = {};
  float m_run[8], l_run[8];
#pragma unroll
  for (int r = 0; r < 8; ++r) { m_run[r] = -3.0e38f; l_run[r] = 0.f; }

  const bf16* kbase = Kb + (size_t)bh * TT * HD;
  const bf16* vbase = Vt + (size_t)bh * HD * TT;

  for (int kb = 0; kb < TT; kb += 32) {
    // ---- load K fragments (2 subtiles x 2 d-chunks), then V fragments ----
    bf16x16 kfr[2][2];
#pragma unroll
    for (int sub = 0; sub < 2; ++sub) {
      const bf16* kp = kbase + (size_t)(kb + sub * 16 + l16) * HD + khalf * 16;
#pragma unroll
      for (int c = 0; c < 2; ++c)
        kfr[sub][c] = *(const bf16x16*)(kp + c * 32);
    }
    bf16x16 vfr[4];
#pragma unroll
    for (int t = 0; t < 4; ++t)
      vfr[t] = *(const bf16x16*)(
          vbase + (size_t)(t * 16 + l16) * TT + kb + khalf * 16);

    // ---- scores S = Q * K^T for 32 keys (two 16-key C-tiles) ----
    f32x8 s[2] = {};
#pragma unroll
    for (int sub = 0; sub < 2; ++sub)
#pragma unroll
      for (int c = 0; c < 2; ++c)
        s[sub] = WMMA_BF16(aq[c], kfr[sub][c], s[sub]);

    // prefetch next K tile into cache (global_prefetch_b8)
    if (kb + 32 < TT)
      __builtin_prefetch(kbase + (size_t)(kb + 32 + l16) * HD, 0, 1);

    // ---- online softmax (row = VGPR slot, col = lane within 16-group) ----
    float corr[8];
#pragma unroll
    for (int r = 0; r < 8; ++r) {
      float tm = fmaxf(s[0][r], s[1][r]);
#pragma unroll
      for (int off = 1; off < 16; off <<= 1)
        tm = fmaxf(tm, __shfl_xor(tm, off, 32));
      const float mnew = fmaxf(m_run[r], tm);
      corr[r] = __expf(m_run[r] - mnew);
      m_run[r] = mnew;
      const float p0 = __expf(s[0][r] - mnew);
      const float p1 = __expf(s[1][r] - mnew);
      s[0][r] = p0; s[1][r] = p1;
      float ps = p0 + p1;
#pragma unroll
      for (int off = 1; off < 16; off <<= 1)
        ps += __shfl_xor(ps, off, 32);
      l_run[r] = l_run[r] * corr[r] + ps;
    }
#pragma unroll
    for (int t = 0; t < 4; ++t)
#pragma unroll
      for (int r = 0; r < 8; ++r) o[t][r] *= corr[r];

    // ---- P: C-layout f32 -> LDS -> A-layout bf16 (16x32) ----
#pragma unroll
    for (int sub = 0; sub < 2; ++sub)
#pragma unroll
      for (int r = 0; r < 8; ++r)
        lds_p[wave][(r + khalf * 8) * 32 + sub * 16 + l16] = (bf16)s[sub][r];
    __syncthreads();
    bf16x16 ap;
    {
      const bf16* lp = &lds_p[wave][l16 * 32 + khalf * 16];
      bf16x8 p0 = *(const bf16x8*)lp;
      bf16x8 p1 = *(const bf16x8*)(lp + 8);
#pragma unroll
      for (int q = 0; q < 8; ++q) { ap[q] = p0[q]; ap[q + 8] = p1[q]; }
    }
    __syncthreads();

    // ---- O += P (16x32) * V (32x64): V^T layout makes B-frags contiguous ----
#pragma unroll
    for (int t = 0; t < 4; ++t)
      o[t] = WMMA_BF16(ap, vfr[t], o[t]);
  }

  // ---- epilogue: 1/l, gate, GroupNorm over this head's 64 channels ----
  float inv_l[8];
#pragma unroll
  for (int r = 0; r < 8; ++r) inv_l[r] = 1.0f / l_run[r];

  float y[4][8];
  float ps[8] = {}, pq[8] = {};
#pragma unroll
  for (int t = 0; t < 4; ++t) {
    const int ch = h * HD + t * 16 + l16;
#pragma unroll
    for (int r = 0; r < 8; ++r) {
      const int row = t0 + r + khalf * 8;
      const float g = G[((size_t)(b * TT + row)) * DD + ch];
      const float v = g * (o[t][r] * inv_l[r]);
      y[t][r] = v;
      ps[r] += v; pq[r] += v * v;
    }
  }
#pragma unroll
  for (int r = 0; r < 8; ++r) {
#pragma unroll
    for (int off = 1; off < 16; off <<= 1) {
      ps[r] += __shfl_xor(ps[r], off, 32);
      pq[r] += __shfl_xor(pq[r], off, 32);
    }
  }
#pragma unroll
  for (int t = 0; t < 4; ++t) {
    const int ch = h * HD + t * 16 + l16;
    const float gm = gamma[ch], bt = beta[ch];
#pragma unroll
    for (int r = 0; r < 8; ++r) {
      const float mean = ps[r] * (1.0f / 64.0f);
      const float var  = pq[r] * (1.0f / 64.0f) - mean * mean;
      const float rs   = rsqrtf(var + EPS_GN);
      const int row = t0 + r + khalf * 8;
      out[((size_t)(b * TT + row)) * DD + ch] = (y[t][r] - mean) * rs * gm + bt;
    }
  }
}

// ---------------------------------------------------------------------------
// Host-side launch.  Workspace layout (bytes):
//   [0,8M)   xb  bf16  (4096x1024)
//   [8,16M)  wb  bf16  (4x1024x1024, q|k|v|g)
//   [16,24M) Qb  bf16  [b,h,t,hd]
//   [24,32M) Kb  bf16  [b,h,t,hd]
//   [32,40M) Vt  bf16  [b,h,hd,t]
//   [40,56M) G   f32   [n,e]
// ---------------------------------------------------------------------------
extern "C" void kernel_launch(void* const* d_in, const int* in_sizes, int n_in,
                              void* d_out, int out_size, void* d_ws, size_t ws_size,
                              hipStream_t stream) {
  const float* x     = (const float*)d_in[0];
  const float* wq    = (const float*)d_in[1];
  const float* bq    = (const float*)d_in[2];
  const float* wk    = (const float*)d_in[3];
  const float* bk    = (const float*)d_in[4];
  const float* wv    = (const float*)d_in[5];
  const float* bv    = (const float*)d_in[6];
  const float* wg    = (const float*)d_in[7];
  const float* bg    = (const float*)d_in[8];
  const float* gamma = (const float*)d_in[9];
  const float* beta  = (const float*)d_in[10];

  char* ws = (char*)d_ws;
  const size_t MB = 1u << 20;
  bf16*  xb = (bf16*)(ws);
  bf16*  wb = (bf16*)(ws + 8 * MB);
  bf16*  Qb = (bf16*)(ws + 16 * MB);
  bf16*  Kb = (bf16*)(ws + 24 * MB);
  bf16*  Vt = (bf16*)(ws + 32 * MB);
  float* G  = (float*)(ws + 40 * MB);

  // 1) downcast x + weights to bf16 (4 elems/thread)
  const size_t total = (size_t)NTOK * DD + 4u * DD * DD;   // 8M elements
  msr_convert_kernel<<<(unsigned)(total / (256 * 4)), 256, 0, stream>>>(
      x, wq, wk, wv, wg, xb, wb);

  // 2) fused QKVG projection GEMM (4096 x 4096 output), 32x64 per wave
  msr_qkvg_gemm_kernel<<<dim3(NTOK / 128, (4 * DD) / 64), 128, 0, stream>>>(
      xb, wb, bq, bk, bv, bg, Qb, Kb, Vt, G);

  // 3) flash attention + gate + GroupNorm, fused
  msr_attn_gn_kernel<<<dim3(TT / 64, BB * HH), 128, 0, stream>>>(
      Qb, Kb, Vt, G, gamma, beta, (float*)d_out);
}
